// Compl_84954453115197
// MI455X (gfx1250) — matrix operand order repaired and checked by reading, not compiled, not run on hardware
//
#include <hip/hip_runtime.h>
#include <stdint.h>

typedef _Float16 f16;
typedef __attribute__((ext_vector_type(16))) _Float16 v16h;
typedef __attribute__((ext_vector_type(8)))  _Float16 v8h;
typedef __attribute__((ext_vector_type(8)))  float    v8f;
typedef __attribute__((vector_size(16)))     int      v4i;

#define LDSTR 136   // LDS row stride in halves (136*2=272B: 16B aligned rows, bank-rotating)

#if defined(__HIP_DEVICE_COMPILE__) && __has_builtin(__builtin_amdgcn_global_load_async_to_lds_b128)
#define ASYNC_LDS 1
#else
#define ASYNC_LDS 0
#endif

// ---------------- WMMA helpers (CDNA5 16x16x32 f16 -> f32) ----------------

__device__ __forceinline__ v8f wmma16(v16h a, v16h b, v8f c) {
    // (neg_a, A, neg_b, B, c_mod, C, reuse_a, reuse_b)
    return __builtin_amdgcn_wmma_f32_16x16x32_f16(false, a, false, b, (short)0, c, false, false);
}

// A-matrix 16x32 f16 layout: lane L holds row M = L&15; halves 0..7 = K (hi*8 .. hi*8+7),
// halves 8..15 = K (16+hi*8 ..), where hi = L>>4.  Two aligned 16B LDS reads.
__device__ __forceinline__ v16h load_A_lds(const f16* base, int stride, int mt, int kbase, int lane) {
    const f16* r = base + (mt * 16 + (lane & 15)) * stride + kbase + ((lane >> 4) << 3);
    union { v16h v; v8h h[2]; } u;
    u.h[0] = *(const v8h*)(r);
    u.h[1] = *(const v8h*)(r + 16);
    return u.v;
}

// B-matrix 32x16 f16 layout: lane L holds column N = L&15; halves 0..7 = K (hi*8 ..),
// halves 8..15 = K (16+hi*8 ..), hi = L>>4. Weights are [K][N=128] f16, gathered with stride 128.
__device__ __forceinline__ v16h load_B_glb(const f16* wt, int nt, int kbase, int lane) {
    const f16* p = wt + (kbase + ((lane >> 4) << 3)) * 128 + nt * 16 + (lane & 15);
    v16h v;
#pragma unroll
    for (int i = 0; i < 8; ++i) {
        v[i] = p[i * 128];
        v[8 + i] = p[(16 + i) * 128];
    }
    return v;
}

// C/D 16x16 f32 layout: lane L -> N = L&15, M = (L>>4)*8 + vgpr.  Fused bias(+ReLU) store.
__device__ __forceinline__ void store_tile(f16* dst, int stride, int mt, int nt, v8f c,
                                           const float* bias, bool do_relu, int lane) {
    int n = nt * 16 + (lane & 15);
    int mbase = mt * 16 + ((lane >> 4) << 3);
    float bv = bias[n];
#pragma unroll
    for (int r = 0; r < 8; ++r) {
        float x = c[r] + bv;
        if (do_relu) x = fmaxf(x, 0.f);
        dst[(mbase + r) * stride + n] = (f16)x;
    }
}

// ---------------- Prep kernel: transpose+cast weights to f16 K-major, fold K-branch ----------------
// ws layout (bytes): fc1t[32][128] @0 (8192) | fc2t[128][128] @8192 (32768) |
//                    vt[128][128] @40960 (32768) | uk[128] @73728 | ck @74240

__global__ void gat_prep(const float* fc1_w, const float* fc2_w, const float* V_w,
                         const float* K_w, const float* QK_w, const float* K_b, const float* QK_b,
                         f16* fc1t, f16* fc2t, f16* vt, float* uk, float* ck) {
    int tid = blockIdx.x * blockDim.x + threadIdx.x;
    if (tid < 4096) {                       // fc1t[k][n] = fc1_w[n][k] (K padded 9->32 with 0)
        int k = tid >> 7, n = tid & 127;
        fc1t[tid] = (f16)((k < 9) ? fc1_w[n * 9 + k] : 0.f);
    } else if (tid < 4096 + 16384) {        // fc2t[k][n] = fc2_w[n][k]
        int i = tid - 4096; int k = i >> 7, n = i & 127;
        fc2t[i] = (f16)fc2_w[n * 128 + k];
    } else if (tid < 4096 + 32768) {        // vt[k][n] = V_w[n][k]
        int i = tid - (4096 + 16384); int k = i >> 7, n = i & 127;
        vt[i] = (f16)V_w[n * 128 + k];
    } else if (tid < 4096 + 32768 + 128) {  // uk = K_w^T wk  (wk = QK_w[0,128:])
        int k = tid - (4096 + 32768);
        float s = 0.f;
        for (int n = 0; n < 128; ++n) s += QK_w[128 + n] * K_w[n * 128 + k];
        uk[k] = s;
    } else if (tid == 4096 + 32768 + 128) { // ck = wk . K_b + QK_b
        float s = QK_b[0];
        for (int n = 0; n < 128; ++n) s += QK_w[128 + n] * K_b[n];
        ck[0] = s;
    }
}

// ---------------- Fused per-batch kernel: 1 workgroup (4 wave32) per batch ----------------

__global__ __launch_bounds__(128) void gat_fused(
    const float* __restrict__ input_xx, const float* __restrict__ input_yy,
    const float* __restrict__ output_xx,
    const float* __restrict__ fc1_b, const float* __restrict__ fc2_b,
    const float* __restrict__ enc_w, const float* __restrict__ enc_b,
    const float* __restrict__ V_b,
    const float* __restrict__ A2N_w, const float* __restrict__ A2N_b,
    const float* __restrict__ pred_w, const float* __restrict__ pred_b,
    const f16* __restrict__ fc1t, const f16* __restrict__ fc2t, const f16* __restrict__ vt,
    const float* __restrict__ uk, const float* __restrict__ ckp,
    float* __restrict__ out) {

    __shared__ __align__(32) f16 nodes1[64 * LDSTR];   // h after fc1 (f16)
    __shared__ __align__(32) f16 nodes2[80 * LDSTR];   // h after fc2 + ho row 64, padded to 80
    __shared__ __align__(32) f16 vbuf[80 * LDSTR];     // v projection; first 4KB aliased as a_xy
    __shared__ __align__(16) float sxy[576];           // async-staged raw inputs: 512 xx + 64 yy
    __shared__ float sk[80];    // k-side scores
    __shared__ float pbuf[80];  // softmax probs
    __shared__ float agv[128];  // p-weighted sum of v rows
    __shared__ float t2[128];   // relu(A2N @ agv + b)

    const int b    = blockIdx.x;
    const int tid  = threadIdx.x;
    const int lane = tid & 31;
    const int w    = tid >> 5;          // wave id (wave32)

    f16* a_xy = vbuf;                   // 64x32 f16, dead before vbuf is written

    // keep the shared weight tiles warm in the near caches
    __builtin_prefetch(fc2t + ((tid * 64) & 16383), 0, 0);
    __builtin_prefetch(vt   + ((tid * 64) & 16383), 0, 0);

    // -------- stage 0: stage raw inputs into LDS (ASYNCcnt path on gfx1250), then pack to f16
#if ASYNC_LDS
    {
        typedef __attribute__((address_space(1))) v4i* gv4p;   // prints as "v4i __device__ *"
        typedef __attribute__((address_space(3))) v4i* sv4p;   // prints as "v4i __shared__ *"
        sv4p s0  = (sv4p)(uint32_t)(uintptr_t)(&sxy[0]);
        gv4p gxx = (gv4p)(uintptr_t)(input_xx + (size_t)b * 512);
        __builtin_amdgcn_global_load_async_to_lds_b128(gxx + tid, s0 + tid, 0, 0);
        if (tid < 16) {
            gv4p gyy = (gv4p)(uintptr_t)(input_yy + (size_t)b * 64);
            __builtin_amdgcn_global_load_async_to_lds_b128(gyy + tid, s0 + 128 + tid, 0, 0);
        }
#if __has_builtin(__builtin_amdgcn_s_wait_asynccnt)
        __builtin_amdgcn_s_wait_asynccnt(0);
#else
        asm volatile("s_wait_asynccnt 0x0" ::: "memory");
#endif
    }
    __syncthreads();
#endif
    for (int i = tid; i < 64 * 32; i += 128) {
        int rr = i >> 5, cc = i & 31;
        float val = 0.f;
#if ASYNC_LDS
        if (cc < 8)       val = sxy[rr * 8 + cc];
        else if (cc == 8) val = sxy[512 + rr];
#else
        if (cc < 8)       val = input_xx[(b * 64 + rr) * 8 + cc];
        else if (cc == 8) val = input_yy[b * 64 + rr];
#endif
        a_xy[rr * 32 + cc] = (f16)val;
    }
    for (int i = tid; i < 15 * LDSTR; i += 128) nodes2[65 * LDSTR + i] = (f16)0.f;
    __syncthreads();

    // -------- stage 1: h1 = relu(xy @ fc1^T + b)   (4 Mtiles x 8 Ntiles, K=32 single step)
    {
        v16h av[4];
#pragma unroll
        for (int mt = 0; mt < 4; ++mt) av[mt] = load_A_lds(a_xy, 32, mt, 0, lane);
#pragma unroll
        for (int tni = 0; tni < 2; ++tni) {
            int nt = 2 * w + tni;
            v16h bf = load_B_glb(fc1t, nt, 0, lane);
            v8f acc[4] = {};
#pragma unroll
            for (int mt = 0; mt < 4; ++mt) acc[mt] = wmma16(av[mt], bf, acc[mt]);
#pragma unroll
            for (int mt = 0; mt < 4; ++mt)
                store_tile(nodes1, LDSTR, mt, nt, acc[mt], fc1_b, true, lane);
        }
    }

    // -------- stage 1.5: ho row (row 64 of nodes2); all threads active, no divergence
    {
        float accv = enc_b[tid];
#pragma unroll
        for (int j = 0; j < 8; ++j) accv += output_xx[b * 8 + j] * enc_w[tid * 8 + j];
        nodes2[64 * LDSTR + tid] = (f16)fmaxf(accv, 0.f);
    }
    __syncthreads();

    // -------- stage 2: h2 = relu(h1 @ fc2^T + b)   (4x8 tiles, 4 K-steps, A reused across 2 Ntiles)
    {
        v8f acc[2][4] = {};
#pragma unroll
        for (int kk = 0; kk < 4; ++kk) {
            v16h av[4];
#pragma unroll
            for (int mt = 0; mt < 4; ++mt) av[mt] = load_A_lds(nodes1, LDSTR, mt, kk * 32, lane);
#pragma unroll
            for (int tni = 0; tni < 2; ++tni) {
                v16h bf = load_B_glb(fc2t, 2 * w + tni, kk * 32, lane);
#pragma unroll
                for (int mt = 0; mt < 4; ++mt) acc[tni][mt] = wmma16(av[mt], bf, acc[tni][mt]);
            }
        }
#pragma unroll
        for (int tni = 0; tni < 2; ++tni)
#pragma unroll
            for (int mt = 0; mt < 4; ++mt)
                store_tile(nodes2, LDSTR, mt, 2 * w + tni, acc[tni][mt], fc2_b, true, lane);
    }
    __syncthreads();

    // -------- stage 3: v = nodes @ V^T + Vb   (5x8 tiles, 4 K-steps, A reused across 2 Ntiles)
    {
        v8f acc[2][5] = {};
#pragma unroll
        for (int kk = 0; kk < 4; ++kk) {
            v16h av[5];
#pragma unroll
            for (int mt = 0; mt < 5; ++mt) av[mt] = load_A_lds(nodes2, LDSTR, mt, kk * 32, lane);
#pragma unroll
            for (int tni = 0; tni < 2; ++tni) {
                v16h bf = load_B_glb(vt, 2 * w + tni, kk * 32, lane);
#pragma unroll
                for (int mt = 0; mt < 5; ++mt) acc[tni][mt] = wmma16(av[mt], bf, acc[tni][mt]);
            }
        }
#pragma unroll
        for (int tni = 0; tni < 2; ++tni)
#pragma unroll
            for (int mt = 0; mt < 5; ++mt)
                store_tile(vbuf, LDSTR, mt, 2 * w + tni, acc[tni][mt], V_b, false, lane);
    }

    // -------- stage 3.5: k-side scores. Additive attention is separable: softmax over j of
    // (sq[i]+sk[j]+c) == softmax(sk) independent of i, so q is dead and attn has identical rows.
    if (tid < 65) {
        float s = ckp[0];
        for (int kx = 0; kx < 128; ++kx) s += (float)nodes2[tid * LDSTR + kx] * uk[kx];
        sk[tid] = s;
    }
    __syncthreads();

    float mx = -3.4e38f;
    for (int j = 0; j < 65; ++j) mx = fmaxf(mx, sk[j]);
    float sum = 0.f;
    for (int j = 0; j < 65; ++j) sum += __expf(sk[j] - mx);
    if (tid < 65) pbuf[tid] = __expf(sk[tid] - mx) / sum;
    __syncthreads();

    // -------- stage 4: agv = sum_j p[j] * v[j,:]   (one column per thread)
    {
        float a = 0.f;
        for (int j = 0; j < 65; ++j) a += pbuf[j] * (float)vbuf[j * LDSTR + tid];
        agv[tid] = a;
    }
    __syncthreads();

    // -------- stage 5: t2 = relu(A2N_w @ agv + b)  (rows of aggre are identical -> row-max == t2)
    {
        float t = A2N_b[tid];
        for (int kx = 0; kx < 128; ++kx) t += A2N_w[tid * 128 + kx] * agv[kx];
        t2[tid] = fmaxf(t, 0.f);
    }
    __syncthreads();

    // -------- stage 6: 2-class head + log_softmax
    if (tid == 0) {
        float l0 = pred_b[0], l1 = pred_b[1];
        for (int kx = 0; kx < 128; ++kx) {
            l0 += pred_w[kx] * t2[kx];
            l1 += pred_w[128 + kx] * t2[kx];
        }
        float m = fmaxf(l0, l1);
        float lse = m + __logf(__expf(l0 - m) + __expf(l1 - m));
        out[b * 2 + 0] = l0 - lse;
        out[b * 2 + 1] = l1 - lse;
    }
}

// ---------------- Launch ----------------

extern "C" void kernel_launch(void* const* d_in, const int* in_sizes, int n_in,
                              void* d_out, int out_size, void* d_ws, size_t ws_size,
                              hipStream_t stream) {
    const float* input_xx = (const float*)d_in[0];
    const float* input_yy = (const float*)d_in[1];
    const float* output_xx = (const float*)d_in[2];
    const float* fc1_w = (const float*)d_in[3];
    const float* fc1_b = (const float*)d_in[4];
    const float* fc2_w = (const float*)d_in[5];
    const float* fc2_b = (const float*)d_in[6];
    const float* enc_w = (const float*)d_in[7];
    const float* enc_b = (const float*)d_in[8];
    // d_in[9], d_in[10]: Q_w, Q_b -- provably dead (softmax shift invariance)
    const float* K_w  = (const float*)d_in[11];
    const float* K_b  = (const float*)d_in[12];
    const float* V_w  = (const float*)d_in[13];
    const float* V_b  = (const float*)d_in[14];
    const float* QK_w = (const float*)d_in[15];
    const float* QK_b = (const float*)d_in[16];
    const float* A2N_w = (const float*)d_in[17];
    const float* A2N_b = (const float*)d_in[18];
    const float* pred_w = (const float*)d_in[19];
    const float* pred_b = (const float*)d_in[20];
    float* out = (float*)d_out;

    char* ws = (char*)d_ws;
    f16*   fc1t = (f16*)(ws);
    f16*   fc2t = (f16*)(ws + 8192);
    f16*   vt   = (f16*)(ws + 40960);
    float* uk   = (float*)(ws + 73728);
    float* ck   = (float*)(ws + 74240);

    gat_prep<<<145, 256, 0, stream>>>(fc1_w, fc2_w, V_w, K_w, QK_w, K_b, QK_b,
                                      fc1t, fc2t, vt, uk, ck);

    gat_fused<<<8192, 128, 0, stream>>>(input_xx, input_yy, output_xx,
                                        fc1_b, fc2_b, enc_w, enc_b, V_b,
                                        A2N_w, A2N_b, pred_w, pred_b,
                                        fc1t, fc2t, vt, uk, ck, out);
}
